// SuperQualityFuzzyModel_86895778333073
// MI455X (gfx1250) — compile-verified, matrix operand order checked by reading, not run on hardware
//
#include <hip/hip_runtime.h>
#include <cmath>

typedef __attribute__((ext_vector_type(16))) _Float16 v16h;
typedef __attribute__((ext_vector_type(8)))  float    v8f;

#define DIV_UP(a, b) (((a) + (b) - 1) / (b))

// ---------------------------------------------------------------------------
// WMMA GEMM:  C[z] = act( alpha * ( scale * (A[z] @ B[z]^T) + bias ) [+ C[z]] )
// A: [M,K] row-major (row stride lda), B: [N,K] row-major (row stride ldb).
// f32 in/out, f16 WMMA compute, f32 accumulate. Block = 256 threads = 8 waves,
// block tile 64(M) x 256(N), wave tile 32x64 (2x4 v_wmma_f32_16x16x32_f16:
// 8 WMMAs per 6 fragment loads). K is always a multiple of 32 in this model.
// ---------------------------------------------------------------------------

// Fragment layout for v_wmma_f32_16x16x32_f16 (A operand; the B operand read
// from a row-major [N,K] buffer uses the identical per-lane pattern, row -> n):
//   lanes 0-15 : row = base+lane,    halves 0..7 = K k0..k0+7,  halves 8..15 = K k0+16..k0+23
//   lanes 16-31: row = base+lane-16, halves 0..7 = K k0+8..+15, halves 8..15 = K k0+24..+31
__device__ __forceinline__ v16h frag_full(const float* __restrict__ base, int row0,
                                          int ld, int k0, int lane) {
  const float* p = base + (long long)(row0 + (lane & 15)) * ld + k0 + ((lane & 16) ? 8 : 0);
  v16h f;
#pragma unroll
  for (int i = 0; i < 8; ++i) {
    f[i]     = (_Float16)p[i];
    f[i + 8] = (_Float16)p[i + 16];
  }
  return f;
}

__device__ __forceinline__ v16h frag_masked(const float* __restrict__ base, int row0,
                                            int Rmax, int ld, int k0, int lane) {
  v16h f;
  int r = row0 + (lane & 15);
  if (r < Rmax) {
    const float* p = base + (long long)r * ld + k0 + ((lane & 16) ? 8 : 0);
#pragma unroll
    for (int i = 0; i < 8; ++i) {
      f[i]     = (_Float16)p[i];
      f[i + 8] = (_Float16)p[i + 16];
    }
  } else {
#pragma unroll
    for (int i = 0; i < 16; ++i) f[i] = (_Float16)0.f;
  }
  return f;
}

template <bool FULL>
__device__ __forceinline__ void gemm_kloop(const float* __restrict__ A,
                                           const float* __restrict__ Bw,
                                           int M, int N, int K, int lda, int ldb,
                                           int m0, int n0, int lane, v8f (&acc)[2][4]) {
  for (int k0 = 0; k0 < K; k0 += 32) {
    if (FULL && k0 + 32 < K) {  // software prefetch next K slice (global_prefetch_b8)
      __builtin_prefetch((const void*)(A  + (long long)(m0 + (lane & 15)) * lda + k0 + 32), 0, 1);
      __builtin_prefetch((const void*)(Bw + (long long)(n0 + (lane & 15)) * ldb + k0 + 32), 0, 1);
      __builtin_prefetch((const void*)(Bw + (long long)(n0 + 32 + (lane & 15)) * ldb + k0 + 32), 0, 1);
    }
    v16h a0, a1, b0, b1, b2, b3;
    if (FULL) {
      a0 = frag_full(A,  m0,      lda, k0, lane);
      a1 = frag_full(A,  m0 + 16, lda, k0, lane);
      b0 = frag_full(Bw, n0,      ldb, k0, lane);
      b1 = frag_full(Bw, n0 + 16, ldb, k0, lane);
      b2 = frag_full(Bw, n0 + 32, ldb, k0, lane);
      b3 = frag_full(Bw, n0 + 48, ldb, k0, lane);
    } else {
      a0 = frag_masked(A,  m0,      M, lda, k0, lane);
      a1 = frag_masked(A,  m0 + 16, M, lda, k0, lane);
      b0 = frag_masked(Bw, n0,      N, ldb, k0, lane);
      b1 = frag_masked(Bw, n0 + 16, N, ldb, k0, lane);
      b2 = frag_masked(Bw, n0 + 32, N, ldb, k0, lane);
      b3 = frag_masked(Bw, n0 + 48, N, ldb, k0, lane);
    }
    acc[0][0] = __builtin_amdgcn_wmma_f32_16x16x32_f16(false, a0, false, b0, (short)0, acc[0][0], false, false);
    acc[0][1] = __builtin_amdgcn_wmma_f32_16x16x32_f16(false, a0, false, b1, (short)0, acc[0][1], false, false);
    acc[0][2] = __builtin_amdgcn_wmma_f32_16x16x32_f16(false, a0, false, b2, (short)0, acc[0][2], false, false);
    acc[0][3] = __builtin_amdgcn_wmma_f32_16x16x32_f16(false, a0, false, b3, (short)0, acc[0][3], false, false);
    acc[1][0] = __builtin_amdgcn_wmma_f32_16x16x32_f16(false, a1, false, b0, (short)0, acc[1][0], false, false);
    acc[1][1] = __builtin_amdgcn_wmma_f32_16x16x32_f16(false, a1, false, b1, (short)0, acc[1][1], false, false);
    acc[1][2] = __builtin_amdgcn_wmma_f32_16x16x32_f16(false, a1, false, b2, (short)0, acc[1][2], false, false);
    acc[1][3] = __builtin_amdgcn_wmma_f32_16x16x32_f16(false, a1, false, b3, (short)0, acc[1][3], false, false);
  }
}

__global__ __launch_bounds__(256) void k_gemm_wmma(
    const float* __restrict__ A, long long sA,
    const float* __restrict__ Bw, long long sB,
    const float* __restrict__ bias,
    float* __restrict__ C, long long sC,
    int M, int N, int K, int lda, int ldb, int ldc,
    const float* __restrict__ alpha_ptr, int alpha_idx,
    float scale, int accumulate, int relu)
{
  A  += (long long)blockIdx.z * sA;
  Bw += (long long)blockIdx.z * sB;
  C  += (long long)blockIdx.z * sC;

  const int lane = threadIdx.x & 31;
  const int wave = threadIdx.x >> 5;
  const int m0 = blockIdx.y * 64  + (wave & 1)  * 32;   // 2 waves down M
  const int n0 = blockIdx.x * 256 + (wave >> 1) * 64;   // 4 waves across N

  v8f acc[2][4] = {};

  // Fast path: this wave's 32x64 tile fully in-range (true for every big GEMM
  // in this model); masked path only for edge tiles (heads: N=2, M=8).
  if (m0 + 32 <= M && n0 + 64 <= N) {
    gemm_kloop<true>(A, Bw, M, N, K, lda, ldb, m0, n0, lane, acc);
  } else {
    gemm_kloop<false>(A, Bw, M, N, K, lda, ldb, m0, n0, lane, acc);
  }

  const float alpha = alpha_ptr ? alpha_ptr[alpha_idx] : 1.0f;

  // C/D layout: VGPR j -> row (lanes 0-15: j, lanes 16-31: 8+j), col = lane&15.
#pragma unroll
  for (int mt = 0; mt < 2; ++mt) {
#pragma unroll
    for (int nt = 0; nt < 4; ++nt) {
      int col  = n0 + nt * 16 + (lane & 15);
      int rowb = m0 + mt * 16 + ((lane & 16) ? 8 : 0);
      if (col < N) {
        float bv = bias ? bias[col] : 0.0f;
#pragma unroll
        for (int j = 0; j < 8; ++j) {
          int row = rowb + j;
          if (row < M) {
            float r = alpha * (acc[mt][nt][j] * scale + bv);
            long long ci = (long long)row * ldc + col;
            if (accumulate) r += C[ci];
            if (relu) r = fmaxf(r, 0.0f);
            C[ci] = r;
          }
        }
      }
    }
  }
}

// ---------------------------------------------------------------------------
// Elementwise / reduction kernels
// ---------------------------------------------------------------------------
__global__ void k_embed_pos(const int* __restrict__ tok, const float* __restrict__ emb,
                            const float* __restrict__ pos, float* __restrict__ X,
                            int Mrows, int S, int d) {
  long long i = (long long)blockIdx.x * blockDim.x + threadIdx.x;
  long long tot = (long long)Mrows * d;
  if (i >= tot) return;
  int m = (int)(i / d), e = (int)(i % d);
  int s = m % S;
  X[i] = emb[(long long)tok[m] * d + e] + pos[(long long)s * d + e];
}

__global__ void k_add_pos(float* __restrict__ X, const float* __restrict__ pos,
                          int Mrows, int S, int d) {
  long long i = (long long)blockIdx.x * blockDim.x + threadIdx.x;
  long long tot = (long long)Mrows * d;
  if (i >= tot) return;
  int m = (int)(i / d), e = (int)(i % d);
  int s = m % S;
  X[i] += pos[(long long)s * d + e];
}

// qkv [B*S, 3d] -> Q[bh][s][dh], K[bh][s][dh], VT[bh][dh][s]
__global__ void k_repack_qkv(const float* __restrict__ qkv, float* __restrict__ Q,
                             float* __restrict__ Km, float* __restrict__ VT,
                             int B, int S, int h, int dh, int d) {
  long long i = (long long)blockIdx.x * blockDim.x + threadIdx.x;
  long long tot = (long long)B * S * d;
  if (i >= tot) return;
  int m = (int)(i / d), c = (int)(i % d);
  int b = m / S, s = m % S;
  int hi = c / dh, e = c % dh;
  long long bh = (long long)b * h + hi;
  const float* row = qkv + (long long)m * (3 * d);
  Q [(bh * S + s) * dh + e] = row[c];
  Km[(bh * S + s) * dh + e] = row[d + c];
  VT[(bh * dh + e) * S + s] = row[2 * d + c];
}

// OP[bh][s][dh] -> O[b][s][h*dh + e]
__global__ void k_merge_heads(const float* __restrict__ OP, float* __restrict__ O,
                              int B, int S, int h, int dh, int d) {
  long long i = (long long)blockIdx.x * blockDim.x + threadIdx.x;
  long long tot = (long long)B * S * d;
  if (i >= tot) return;
  int m = (int)(i / d), c = (int)(i % d);
  int b = m / S, s = m % S;
  int hi = c / dh, e = c % dh;
  O[i] = OP[(((long long)b * h + hi) * S + s) * dh + e];
}

__global__ void k_softmax_rows(float* __restrict__ P, int cols) {
  __shared__ float sm[256];
  float* row = P + (long long)blockIdx.x * cols;
  int t = threadIdx.x;
  float v = (t < cols) ? row[t] : -3.4e38f;
  sm[t] = v; __syncthreads();
  for (int s = 128; s > 0; s >>= 1) { if (t < s) sm[t] = fmaxf(sm[t], sm[t + s]); __syncthreads(); }
  float mx = sm[0]; __syncthreads();
  float e = (t < cols) ? expf(v - mx) : 0.0f;
  sm[t] = e; __syncthreads();
  for (int s = 128; s > 0; s >>= 1) { if (t < s) sm[t] += sm[t + s]; __syncthreads(); }
  float inv = 1.0f / sm[0];
  if (t < cols) row[t] = e * inv;
}

// X = LN(ACC + X) * g + b   (row length d <= 1024, blockDim = 256)
__global__ void k_ln_res(const float* __restrict__ ACC, float* __restrict__ X,
                         const float* __restrict__ g, const float* __restrict__ b, int d) {
  __shared__ float sm[256];
  long long base = (long long)blockIdx.x * d;
  int t = threadIdx.x;
  const int nper = DIV_UP(d, 256);  // <= 4
  float r[4];
  float s = 0.0f;
  for (int j = 0; j < nper; ++j) {
    int c = t + j * 256;
    float v = 0.0f;
    if (c < d) v = ACC[base + c] + X[base + c];
    r[j] = v; s += v;
  }
  sm[t] = s; __syncthreads();
  for (int st = 128; st > 0; st >>= 1) { if (t < st) sm[t] += sm[t + st]; __syncthreads(); }
  float mu = sm[0] / d; __syncthreads();
  float vs = 0.0f;
  for (int j = 0; j < nper; ++j) {
    int c = t + j * 256;
    if (c < d) { float dv = r[j] - mu; vs += dv * dv; }
  }
  sm[t] = vs; __syncthreads();
  for (int st = 128; st > 0; st >>= 1) { if (t < st) sm[t] += sm[t + st]; __syncthreads(); }
  float inv = rsqrtf(sm[0] / d + 1e-5f);
  for (int j = 0; j < nper; ++j) {
    int c = t + j * 256;
    if (c < d) X[base + c] = (r[j] - mu) * inv * g[c] + b[c];
  }
}

__global__ void k_mean_s(const float* __restrict__ Xin, float* __restrict__ out,
                         int B, int S, int d) {
  int i = blockIdx.x * blockDim.x + threadIdx.x;
  if (i >= B * d) return;
  int b = i / d, e = i % d;
  float s = 0.0f;
  for (int ss = 0; ss < S; ++ss) s += Xin[((long long)b * S + ss) * d + e];
  out[i] = s / (float)S;
}

__global__ void k_concat2(const float* __restrict__ A, const float* __restrict__ Bp,
                          float* __restrict__ O, int rows, int dA, int dB) {
  int i = blockIdx.x * blockDim.x + threadIdx.x;
  int dO = dA + dB;
  if (i >= rows * dO) return;
  int r = i / dO, c = i % dO;
  O[i] = (c < dA) ? A[r * dA + c] : Bp[r * dB + (c - dA)];
}

__global__ void k_softmax_vec(const float* __restrict__ in, float* __restrict__ out, int n) {
  if (threadIdx.x == 0 && blockIdx.x == 0) {
    float m = in[0];
    for (int i = 1; i < n; ++i) m = fmaxf(m, in[i]);
    float s = 0.0f;
    for (int i = 0; i < n; ++i) s += expf(in[i] - m);
    for (int i = 0; i < n; ++i) out[i] = expf(in[i] - m) / s;
  }
}

// logits [B,2] -> d_out: [logits(2B) | probs(2B)]
__global__ void k_final(const float* __restrict__ L, float* __restrict__ out, int B) {
  int b = threadIdx.x;
  if (b >= B) return;
  float l0 = L[b * 2], l1 = L[b * 2 + 1];
  out[b * 2] = l0; out[b * 2 + 1] = l1;
  float m = fmaxf(l0, l1);
  float e0 = expf(l0 - m), e1 = expf(l1 - m);
  float s = e0 + e1;
  out[2 * B + b * 2] = e0 / s; out[2 * B + b * 2 + 1] = e1 / s;
}

// ---------------------------------------------------------------------------
// Host-side orchestration
// ---------------------------------------------------------------------------
struct MhaP { const float *w_in, *b_in, *w_out, *b_out; };
struct MhaScr { float *qkv, *Q, *K, *VT, *P, *OP, *O; };

static void launch_gemm(hipStream_t st,
                        const float* A, long long sA, const float* Bw, long long sB,
                        const float* bias, float* C, long long sC,
                        int M, int N, int K, int lda, int ldb, int ldc,
                        const float* alpha, int aidx, float scale, int acc, int relu, int batch) {
  dim3 grid(DIV_UP(N, 256), DIV_UP(M, 64), batch);
  k_gemm_wmma<<<grid, 256, 0, st>>>(A, sA, Bw, sB, bias, C, sC, M, N, K, lda, ldb, ldc,
                                    alpha, aidx, scale, acc, relu);
}

static void run_mha_self(hipStream_t st, const float* Xin, int B, int S, int d, int h,
                         const MhaP& p, float* Cout, const float* alpha, int aidx, int acc,
                         const MhaScr& sc) {
  const int M = B * S, dh = d / h, bh = B * h;
  // qkv = X @ w_in^T + b_in
  launch_gemm(st, Xin, 0, p.w_in, 0, p.b_in, sc.qkv, 0, M, 3 * d, d, d, d, 3 * d,
              nullptr, 0, 1.0f, 0, 0, 1);
  long long tot = (long long)M * d;
  k_repack_qkv<<<dim3((unsigned)DIV_UP(tot, 256)), 256, 0, st>>>(sc.qkv, sc.Q, sc.K, sc.VT, B, S, h, dh, d);
  // P = softmax(Q @ K^T / sqrt(dh))
  launch_gemm(st, sc.Q, (long long)S * dh, sc.K, (long long)S * dh, nullptr, sc.P, (long long)S * S,
              S, S, dh, dh, dh, S, nullptr, 0, 1.0f / sqrtf((float)dh), 0, 0, bh);
  k_softmax_rows<<<bh * S, 256, 0, st>>>(sc.P, S);
  // OP = P @ V  (V^T stored [dh, S] -> same [N,K] loader)
  launch_gemm(st, sc.P, (long long)S * S, sc.VT, (long long)dh * S, nullptr, sc.OP, (long long)S * dh,
              S, dh, S, S, S, dh, nullptr, 0, 1.0f, 0, 0, bh);
  k_merge_heads<<<dim3((unsigned)DIV_UP(tot, 256)), 256, 0, st>>>(sc.OP, sc.O, B, S, h, dh, d);
  // Cout (+)= alpha * (O @ w_out^T + b_out)
  launch_gemm(st, sc.O, 0, p.w_out, 0, p.b_out, Cout, 0, M, d, d, d, d, d,
              alpha, aidx, 1.0f, acc, 0, 1);
}

extern "C" void kernel_launch(void* const* d_in, const int* in_sizes, int n_in,
                              void* d_out, int out_size, void* d_ws, size_t ws_size,
                              hipStream_t stream) {
  (void)in_sizes; (void)n_in; (void)out_size; (void)ws_size;
  const int B = 8, S = 256, M = B * S;

  // -------- walk the flattened params (setup_inputs insertion order) --------
  int gi = 0;
  auto NXT = [&]() -> const float* { return (const float*)d_in[gi++]; };
  const int*   tok  = (const int*)d_in[gi++];
  const float* imgf = NXT();

  // text
  const float* t_emb = NXT();
  const float* t_pos = NXT();
  MhaP t_mha[2][4]; const float *t_ens[2], *t_ln_g[2], *t_ln_b[2];
  for (int l = 0; l < 2; ++l) {
    for (int i = 0; i < 4; ++i) { t_mha[l][i].w_in = NXT(); t_mha[l][i].b_in = NXT();
                                  t_mha[l][i].w_out = NXT(); t_mha[l][i].b_out = NXT(); }
    NXT(); NXT(); NXT(); NXT(); NXT();   // centers, widths, fweights, temp, scale (dead)
    t_ens[l] = NXT(); t_ln_g[l] = NXT(); t_ln_b[l] = NXT();
  }
  const float *t_fe_w[3], *t_fe_b[3];
  for (int j = 0; j < 3; ++j) { t_fe_w[j] = NXT(); t_fe_b[j] = NXT(); }
  MhaP t_pool; t_pool.w_in = NXT(); t_pool.b_in = NXT(); t_pool.w_out = NXT(); t_pool.b_out = NXT();

  // image
  const float* i_proj_w = NXT(); const float* i_proj_b = NXT();
  const float* i_pos = NXT();
  MhaP i_mha[2][4]; const float *i_ens[2], *i_ln_g[2], *i_ln_b[2];
  for (int l = 0; l < 2; ++l) {
    for (int i = 0; i < 4; ++i) { i_mha[l][i].w_in = NXT(); i_mha[l][i].b_in = NXT();
                                  i_mha[l][i].w_out = NXT(); i_mha[l][i].b_out = NXT(); }
    NXT(); NXT(); NXT(); NXT(); NXT();
    i_ens[l] = NXT(); i_ln_g[l] = NXT(); i_ln_b[l] = NXT();
  }
  const float *i_fe_w[3], *i_fe_b[3];
  for (int j = 0; j < 3; ++j) { i_fe_w[j] = NXT(); i_fe_b[j] = NXT(); }
  MhaP i_pool; i_pool.w_in = NXT(); i_pool.b_in = NXT(); i_pool.w_out = NXT(); i_pool.b_out = NXT();

  // cross
  MhaP c_mha[4];
  for (int i = 0; i < 4; ++i) { c_mha[i].w_in = NXT(); c_mha[i].b_in = NXT();
                                c_mha[i].w_out = NXT(); c_mha[i].b_out = NXT(); }
  const float* c_ens = NXT();
  const float *cf_w[3], *cf_b[3];
  for (int j = 0; j < 3; ++j) { cf_w[j] = NXT(); cf_b[j] = NXT(); }

  // heads
  const float *hw[3][4], *hb[3][4];
  for (int j = 0; j < 3; ++j)
    for (int k = 0; k < 4; ++k) { hw[j][k] = NXT(); hb[j][k] = NXT(); }
  const float* cls_ens = NXT();

  // -------- workspace layout (bump allocator) --------
  char* wsb = (char*)d_ws; size_t off = 0;
  auto allocf = [&](size_t elems) -> float* {
    float* p = (float*)(wsb + off);
    off += ((elems * sizeof(float)) + 255) & ~(size_t)255;
    return p;
  };
  MhaScr sc;
  sc.qkv = allocf((size_t)M * 3072);
  sc.Q   = allocf((size_t)M * 1024);
  sc.K   = allocf((size_t)M * 1024);
  sc.VT  = allocf((size_t)M * 1024);
  sc.P   = allocf((size_t)64 * S * S);     // up to B*h = 64 heads
  sc.OP  = allocf((size_t)M * 1024);
  sc.O   = allocf((size_t)M * 1024);
  float* X    = allocf((size_t)M * 1024);
  float* ACC  = allocf((size_t)M * 1024);
  float* T1   = allocf((size_t)M * 1024);
  float* T2   = allocf((size_t)M * 1024);
  float* ewb  = allocf(64);                // [0]=textL0 [4]=textL1 [8]=imgL0 [12]=imgL1 [16]=cross [20]=cls
  float* pooled_t  = allocf(B * 1024);
  float* pooled_im = allocf(B * 1024);
  float* ta8   = allocf(B * 1024);
  float* v8t   = allocf(B * 1024);
  float* fused = allocf(B * 2048);
  float* f1    = allocf(B * 1024);
  float* f2    = allocf(B * 1024);
  float* f3    = allocf(B * 1024);
  float* hb1   = allocf(B * 1024);
  float* hb2   = allocf(B * 512);
  float* hb3   = allocf(B * 256);
  float* Lg    = allocf(B * 2);

  // -------- ensemble softmax weights --------
  k_softmax_vec<<<1, 32, 0, stream>>>(t_ens[0], ewb + 0, 4);
  k_softmax_vec<<<1, 32, 0, stream>>>(t_ens[1], ewb + 4, 4);
  k_softmax_vec<<<1, 32, 0, stream>>>(i_ens[0], ewb + 8, 4);
  k_softmax_vec<<<1, 32, 0, stream>>>(i_ens[1], ewb + 12, 4);
  k_softmax_vec<<<1, 32, 0, stream>>>(c_ens,    ewb + 16, 4);
  k_softmax_vec<<<1, 32, 0, stream>>>(cls_ens,  ewb + 20, 3);

  // -------- text encoder (d=768, h=4) --------
  {
    const int d = 768, h = 4;
    long long tot = (long long)M * d;
    k_embed_pos<<<dim3((unsigned)DIV_UP(tot, 256)), 256, 0, stream>>>(tok, t_emb, t_pos, X, M, S, d);
    for (int l = 0; l < 2; ++l) {
      for (int i = 0; i < 4; ++i)
        run_mha_self(stream, X, B, S, d, h, t_mha[l][i], ACC, ewb + l * 4, i, (i > 0), sc);
      k_ln_res<<<M, 256, 0, stream>>>(ACC, X, t_ln_g[l], t_ln_b[l], d);
    }
    // fe MLP: 768 -> 1024 -> 1024 -> 1024, all ReLU
    launch_gemm(stream, X,  0, t_fe_w[0], 0, t_fe_b[0], T1, 0, M, 1024, 768,  768,  768,  1024, nullptr, 0, 1.0f, 0, 1, 1);
    launch_gemm(stream, T1, 0, t_fe_w[1], 0, t_fe_b[1], T2, 0, M, 1024, 1024, 1024, 1024, 1024, nullptr, 0, 1.0f, 0, 1, 1);
    launch_gemm(stream, T2, 0, t_fe_w[2], 0, t_fe_b[2], T1, 0, M, 1024, 1024, 1024, 1024, 1024, nullptr, 0, 1.0f, 0, 1, 1);
    // pool MHA (d=1024, h=8), then mean over S
    run_mha_self(stream, T1, B, S, 1024, 8, t_pool, ACC, nullptr, 0, 0, sc);
    k_mean_s<<<DIV_UP(B * 1024, 256), 256, 0, stream>>>(ACC, pooled_t, B, S, 1024);
  }

  // -------- image encoder (d=1024, h=4) --------
  {
    const int d = 1024, h = 4;
    long long tot = (long long)M * d;
    // projection 2048 -> 1024, then +pos
    launch_gemm(stream, imgf, 0, i_proj_w, 0, i_proj_b, X, 0, M, 1024, 2048, 2048, 2048, 1024, nullptr, 0, 1.0f, 0, 0, 1);
    k_add_pos<<<dim3((unsigned)DIV_UP(tot, 256)), 256, 0, stream>>>(X, i_pos, M, S, d);
    for (int l = 0; l < 2; ++l) {
      for (int i = 0; i < 4; ++i)
        run_mha_self(stream, X, B, S, d, h, i_mha[l][i], ACC, ewb + 8 + l * 4, i, (i > 0), sc);
      k_ln_res<<<M, 256, 0, stream>>>(ACC, X, i_ln_g[l], i_ln_b[l], d);
    }
    launch_gemm(stream, X,  0, i_fe_w[0], 0, i_fe_b[0], T1, 0, M, 1024, 1024, 1024, 1024, 1024, nullptr, 0, 1.0f, 0, 1, 1);
    launch_gemm(stream, T1, 0, i_fe_w[1], 0, i_fe_b[1], T2, 0, M, 1024, 1024, 1024, 1024, 1024, nullptr, 0, 1.0f, 0, 1, 1);
    launch_gemm(stream, T2, 0, i_fe_w[2], 0, i_fe_b[2], T1, 0, M, 1024, 1024, 1024, 1024, 1024, nullptr, 0, 1.0f, 0, 1, 1);
    run_mha_self(stream, T1, B, S, 1024, 8, i_pool, ACC, nullptr, 0, 0, sc);
    k_mean_s<<<DIV_UP(B * 1024, 256), 256, 0, stream>>>(ACC, pooled_im, B, S, 1024);
  }

  // -------- cross attention (Sq=Sk=1 => softmax over one key => o = v) --------
  // ta = sum_i ew[i] * ((im @ Wv_i^T + bv_i) @ Wo_i^T + bo_i)
  {
    const int d = 1024;
    for (int i = 0; i < 4; ++i) {
      const float* wv = c_mha[i].w_in + (long long)2 * d * d;   // rows [2d, 3d)
      const float* bv = c_mha[i].b_in + 2 * d;
      launch_gemm(stream, pooled_im, 0, wv, 0, bv, v8t, 0, B, d, d, d, d, d, nullptr, 0, 1.0f, 0, 0, 1);
      launch_gemm(stream, v8t, 0, c_mha[i].w_out, 0, c_mha[i].b_out, ta8, 0, B, d, d, d, d, d,
                  ewb + 16, i, 1.0f, (i > 0), 0, 1);
    }
    k_concat2<<<DIV_UP(B * 2048, 256), 256, 0, stream>>>(ta8, pooled_im, fused, B, 1024, 1024);
    // fusion MLP: 2048 -> 1024 -> 1024 -> 1024, all ReLU
    launch_gemm(stream, fused, 0, cf_w[0], 0, cf_b[0], f1, 0, B, 1024, 2048, 2048, 2048, 1024, nullptr, 0, 1.0f, 0, 1, 1);
    launch_gemm(stream, f1,    0, cf_w[1], 0, cf_b[1], f2, 0, B, 1024, 1024, 1024, 1024, 1024, nullptr, 0, 1.0f, 0, 1, 1);
    launch_gemm(stream, f2,    0, cf_w[2], 0, cf_b[2], f3, 0, B, 1024, 1024, 1024, 1024, 1024, nullptr, 0, 1.0f, 0, 1, 1);
    (void)pooled_t;  // reference's t enters cross only via _mha(t, im, ...) whose q/k path is dead (o=v)
  }

  // -------- classification heads (3-way ensemble) --------
  for (int j = 0; j < 3; ++j) {
    launch_gemm(stream, f3,  0, hw[j][0], 0, hb[j][0], hb1, 0, B, 1024, 1024, 1024, 1024, 1024, nullptr, 0, 1.0f, 0, 1, 1);
    launch_gemm(stream, hb1, 0, hw[j][1], 0, hb[j][1], hb2, 0, B, 512,  1024, 1024, 1024, 512,  nullptr, 0, 1.0f, 0, 1, 1);
    launch_gemm(stream, hb2, 0, hw[j][2], 0, hb[j][2], hb3, 0, B, 256,  512,  512,  512,  256,  nullptr, 0, 1.0f, 0, 1, 1);
    launch_gemm(stream, hb3, 0, hw[j][3], 0, hb[j][3], Lg,  0, B, 2,    256,  256,  256,  2,    ewb + 20, j, 1.0f, (j > 0), 0, 1);
  }

  k_final<<<1, 32, 0, stream>>>(Lg, (float*)d_out, B);
}
